// AttentionToVec_30734785970706
// MI455X (gfx1250) — matrix operand order repaired and checked by reading, not compiled, not run on hardware
//
#include <hip/hip_runtime.h>
#include <math.h>

// ---------------------------------------------------------------------------
// AttentionToVec, restructured for MI455X (gfx1250):
//   score[b,n,h] = x[b,n,:] . wq[:,h] + c[h]        (wq = W_k_head @ q_h)
//   a = softmax_n(score)  (mask applied)
//   y[b,h,:] = sum_n a[b,h,n] * x[b,n,:]            (a^T @ x, per batch)
//   sampled[b,h*64+d] = y[b,h,:] . W_v[:,h*64+d] + b_v
//   out = sampled + MLP(sampled)
// Two streaming passes over x (134 MB each; x fits in 192 MB L2 so pass 2
// mostly hits L2). Both big passes use v_wmma_f32_16x16x32_bf16.
// Pass 2 is K-split 8-ways per block (4096 waves) to extract L2 bandwidth.
// ---------------------------------------------------------------------------

typedef __attribute__((ext_vector_type(16))) __bf16 bf16x16;
typedef __attribute__((ext_vector_type(8)))  __bf16 bf16x8;
typedef __attribute__((ext_vector_type(8)))  float  f32x8;

#define NEMB   1024
#define NOUT   1024
#define NHEAD  16
#define HDIM   64
#define SEQ    4096
#define BATCH  8
#define HID    4096   // MULT * NOUT

// ---------------- Kernel A: wq[h][k] = sum_d W_kv[k, h*64+d]*q[h,d]; c[h] ----
__global__ void prep_wq_kernel(const float* __restrict__ Wkv,
                               const float* __restrict__ bkv,
                               const float* __restrict__ query,
                               __bf16* __restrict__ wqT,   // [NHEAD][NEMB]
                               float*  __restrict__ cvec)  // [NHEAD]
{
    int idx = blockIdx.x * blockDim.x + threadIdx.x;   // 0..16383
    int h = idx >> 10;
    int k = idx & 1023;
    const float* q = query + h * HDIM;
    const float* w = Wkv + (size_t)k * (2 * NOUT) + h * HDIM;
    float acc = 0.f;
#pragma unroll 8
    for (int d = 0; d < HDIM; ++d) acc += w[d] * q[d];
    wqT[h * NEMB + k] = (__bf16)acc;

    if (blockIdx.x == 0 && threadIdx.x < NHEAD) {
        int hh = threadIdx.x;
        float cc = 0.f;
        for (int d = 0; d < HDIM; ++d) cc += bkv[hh * HDIM + d] * query[hh * HDIM + d];
        cvec[hh] = cc;
    }
}

// ---------------- Kernel B: scores via WMMA  [32768x1024]x[1024x16] ---------
// One wave handles a 16-row tile of x; K-loop over 1024 in steps of 32.
__global__ void __launch_bounds__(256)
score_kernel(const float* __restrict__ x,
             const __bf16* __restrict__ wqT,
             const float* __restrict__ cvec,
             float* __restrict__ sbuf)            // [B][H][SEQ]
{
    int wave = blockIdx.x * 8 + (threadIdx.x >> 5);  // 0..2047
    int lane = threadIdx.x & 31;
    int rowbase = wave * 16;                          // flattened (b*SEQ+n)
    int m  = lane & 15;
    int hi = lane >> 4;
    int selA = hi ? 16 : 0;   // A: lanes 16-31 carry K=16..31
    int kbB  = hi ? 16 : 0;   // B: lanes 16-31 carry K=16..31

    const float*  xr   = x + (size_t)(rowbase + m) * NEMB;
    const __bf16* wrow = wqT + m * NEMB;              // B column n = lane&15

    f32x8 acc = {0.f, 0.f, 0.f, 0.f, 0.f, 0.f, 0.f, 0.f};
    for (int kk = 0; kk < NEMB; kk += 32) {
        __builtin_prefetch(xr + kk + 128, 0, 0);      // global_prefetch_b8
        float4 f0 = *(const float4*)(xr + kk + selA);
        float4 f1 = *(const float4*)(xr + kk + selA + 4);
        float4 f2 = *(const float4*)(xr + kk + selA + 8);
        float4 f3 = *(const float4*)(xr + kk + selA + 12);
        bf16x16 A;
        A[0]=(__bf16)f0.x; A[1]=(__bf16)f0.y; A[2]=(__bf16)f0.z; A[3]=(__bf16)f0.w;
        A[4]=(__bf16)f1.x; A[5]=(__bf16)f1.y; A[6]=(__bf16)f1.z; A[7]=(__bf16)f1.w;
        A[8]=(__bf16)f2.x; A[9]=(__bf16)f2.y; A[10]=(__bf16)f2.z; A[11]=(__bf16)f2.w;
        A[12]=(__bf16)f3.x; A[13]=(__bf16)f3.y; A[14]=(__bf16)f3.z; A[15]=(__bf16)f3.w;

        bf16x16 B = *(const bf16x16*)(wrow + kk + kbB);  // 16 consecutive K

        acc = __builtin_amdgcn_wmma_f32_16x16x32_bf16(
                  false, A, false, B, (short)0, acc, false, false);
    }

    // D layout: lane = N (head); VGPR r = M row r (lanes 0-15) / r+8 (16-31)
    int h    = lane & 15;
    int mo   = hi ? 8 : 0;
    int b    = rowbase >> 12;       // / SEQ
    int nloc = rowbase & (SEQ - 1);
    float bias = cvec[h];
#pragma unroll
    for (int r = 0; r < 8; ++r) {
        int n = nloc + mo + r;
        sbuf[((size_t)(b * NHEAD + h)) * SEQ + n] = acc[r] + bias;
    }
}

// ---------------- Kernel C: softmax over SEQ per (b,h); emit bf16 weights ---
__global__ void __launch_bounds__(256)
softmax_kernel(const float* __restrict__ sbuf,
               const unsigned char* __restrict__ mask,  // [B][SEQ], all-false here
               __bf16* __restrict__ abuf)               // [B][H][SEQ]
{
    int row = blockIdx.x;            // b*16 + h
    int b   = row >> 4;
    const float* sr = sbuf + (size_t)row * SEQ;
    __shared__ float red[256];

    float mx = -INFINITY;
    for (int n = threadIdx.x; n < SEQ; n += 256) {
        float v = sr[n];
        if (mask[b * SEQ + n]) v = -INFINITY;
        mx = fmaxf(mx, v);
    }
    red[threadIdx.x] = mx; __syncthreads();
    for (int st = 128; st > 0; st >>= 1) {
        if (threadIdx.x < st) red[threadIdx.x] = fmaxf(red[threadIdx.x], red[threadIdx.x + st]);
        __syncthreads();
    }
    mx = red[0]; __syncthreads();

    float sum = 0.f;
    for (int n = threadIdx.x; n < SEQ; n += 256) {
        float v = sr[n];
        if (mask[b * SEQ + n]) v = -INFINITY;
        sum += __expf(v - mx);
    }
    red[threadIdx.x] = sum; __syncthreads();
    for (int st = 128; st > 0; st >>= 1) {
        if (threadIdx.x < st) red[threadIdx.x] += red[threadIdx.x + st];
        __syncthreads();
    }
    float inv = 1.f / red[0];

    for (int n = threadIdx.x; n < SEQ; n += 256) {
        float v = sr[n];
        if (mask[b * SEQ + n]) v = -INFINITY;
        abuf[(size_t)row * SEQ + n] = (__bf16)(__expf(v - mx) * inv);
    }
}

// ---------------- Kernel D: y[b,h,:] = a^T @ x via WMMA, 8-way K-split ------
// Block = 8 waves on one (b, 16-wide column tile); each wave accumulates a
// 512-deep K chunk, then a flat per-thread LDS sum combines the 8 partial fragments.
__global__ void __launch_bounds__(256)
weighted_sum_kernel(const float* __restrict__ x,
                    const __bf16* __restrict__ abuf,
                    float* __restrict__ ybuf)            // [B][H][NEMB]
{
    __shared__ float part[8 * 256];                      // [wave][lane*8+r], 8 KB
    int tid  = threadIdx.x;
    int wv   = tid >> 5;
    int lane = tid & 31;
    int b    = blockIdx.x >> 6;
    int ct   = blockIdx.x & 63;
    int col0 = ct * 16;
    int m    = lane & 15;
    int hi   = lane >> 4;
    int selA = hi ? 16 : 0;
    int kbB  = hi ? 16 : 0;

    const __bf16* ar = abuf + (size_t)(b * NHEAD + m) * SEQ;      // A row = head m
    const float*  xb = x + (size_t)b * SEQ * NEMB + col0 + m;     // B col = col0+m

    f32x8 acc = {0.f, 0.f, 0.f, 0.f, 0.f, 0.f, 0.f, 0.f};
    int k0 = wv * (SEQ / 8);                              // 512-deep chunk
    for (int kk = k0; kk < k0 + (SEQ / 8); kk += 32) {
        bf16x8 a0 = *(const bf16x8*)(ar + kk + selA);
        bf16x8 a1 = *(const bf16x8*)(ar + kk + 8 + selA);
        bf16x16 A;
#pragma unroll
        for (int t = 0; t < 8; ++t) { A[t] = a0[t]; A[8 + t] = a1[t]; }

        const float* xk = xb + (size_t)(kk + kbB) * NEMB;
        bf16x16 B;
#pragma unroll
        for (int t = 0; t < 16; ++t) B[t] = (__bf16)xk[(size_t)t * NEMB];

        acc = __builtin_amdgcn_wmma_f32_16x16x32_bf16(
                  false, A, false, B, (short)0, acc, false, false);
    }

#pragma unroll
    for (int r = 0; r < 8; ++r) part[wv * 256 + lane * 8 + r] = acc[r];
    __syncthreads();

    // One output element per thread: sum the 8 wave-partials.
    float s = 0.f;
#pragma unroll
    for (int w = 0; w < 8; ++w) s += part[w * 256 + tid];
    int L    = tid >> 3;                 // source lane of the fragment element
    int r    = tid & 7;                  // fragment register index
    int col  = col0 + (L & 15);
    int head = r + ((L >> 4) ? 8 : 0);
    ybuf[((size_t)(b * NHEAD + head)) * NEMB + col] = s;
}

// ---------------- Kernel E: sampled[b,h*64+d] = y[b,h,:] . W_v[:,h*64+d] ----
// 256 threads = 4 groups x 64 outputs; each group covers a 256-deep i chunk.
__global__ void __launch_bounds__(256)
sampled_kernel(const float* __restrict__ ybuf,
               const float* __restrict__ Wkv,
               const float* __restrict__ bkv,
               float* __restrict__ samp)   // [B][NOUT]
{
    __shared__ float red[256];
    int bh = blockIdx.x;                 // b*16 + h
    int b = bh >> 4, h = bh & 15;
    int d = threadIdx.x & 63;
    int g = threadIdx.x >> 6;            // 0..3
    const float* yr = ybuf + (size_t)bh * NEMB;
    int col = NOUT + h * HDIM + d;       // W_v half of W_kv
    float acc = 0.f;
    for (int i = g * 256; i < g * 256 + 256; ++i)
        acc += yr[i] * Wkv[(size_t)i * (2 * NOUT) + col];
    red[threadIdx.x] = acc;
    __syncthreads();
    if (threadIdx.x < 64) {
        float s = bkv[col] + red[d] + red[64 + d] + red[128 + d] + red[192 + d];
        samp[b * NOUT + h * HDIM + d] = s;
    }
}

// ---------------- Kernel F1: h = gelu(sampled @ W1 + b1) --------------------
__global__ void __launch_bounds__(256)
mlp1_kernel(const float* __restrict__ samp, const float* __restrict__ W1,
            const float* __restrict__ b1, float* __restrict__ hbuf)
{
    __shared__ float sld[NOUT];
    int b  = blockIdx.x >> 4;
    int jt = blockIdx.x & 15;
    for (int i = threadIdx.x; i < NOUT; i += 256) sld[i] = samp[b * NOUT + i];
    __syncthreads();
    int j = jt * 256 + threadIdx.x;
    float acc = b1[j];
    for (int i = 0; i < NOUT; ++i) acc += sld[i] * W1[(size_t)i * HID + j];
    float t = 0.7978845608028654f * (acc + 0.044715f * acc * acc * acc);
    hbuf[(size_t)b * HID + j] = 0.5f * acc * (1.f + tanhf(t));
}

// ---------------- Kernel F2: out = sampled + h @ W2 + b2 --------------------
__global__ void __launch_bounds__(256)
mlp2_kernel(const float* __restrict__ hbuf, const float* __restrict__ W2,
            const float* __restrict__ b2, const float* __restrict__ samp,
            float* __restrict__ out)
{
    __shared__ float hld[HID];           // 16 KB
    int b  = blockIdx.x >> 2;
    int jt = blockIdx.x & 3;
    for (int i = threadIdx.x; i < HID; i += 256) hld[i] = hbuf[(size_t)b * HID + i];
    __syncthreads();
    int j = jt * 256 + threadIdx.x;
    float acc = b2[j];
    for (int i = 0; i < HID; ++i) acc += hld[i] * W2[(size_t)i * NOUT + j];
    out[b * NOUT + j] = samp[b * NOUT + j] + acc;
}

// ---------------------------------------------------------------------------
extern "C" void kernel_launch(void* const* d_in, const int* in_sizes, int n_in,
                              void* d_out, int out_size, void* d_ws, size_t ws_size,
                              hipStream_t stream)
{
    const float*         x     = (const float*)d_in[0];
    const unsigned char* mask  = (const unsigned char*)d_in[1];
    const float*         Wkv   = (const float*)d_in[2];
    const float*         bkv   = (const float*)d_in[3];
    const float*         query = (const float*)d_in[4];
    const float*         W1    = (const float*)d_in[5];
    const float*         b1    = (const float*)d_in[6];
    const float*         W2    = (const float*)d_in[7];
    const float*         b2    = (const float*)d_in[8];
    float* out = (float*)d_out;

    // Workspace layout (~3.9 MB, all offsets 256B-aligned)
    char* ws = (char*)d_ws;
    __bf16* wqT  = (__bf16*)(ws + 0);                         // 32 KB
    float*  cvec = (float*)(ws + 32768);                      // 64 B
    float*  sbuf = (float*)(ws + 33280);                      // 2 MB
    __bf16* abuf = (__bf16*)(ws + 33280 + 2097152);           // 1 MB
    float*  ybuf = (float*)(ws + 33280 + 2097152 + 1048576);  // 512 KB
    float*  samp = (float*)(ws + 33280 + 2097152 + 1048576 + 524288);          // 32 KB
    float*  hbuf = (float*)(ws + 33280 + 2097152 + 1048576 + 524288 + 32768);  // 128 KB

    prep_wq_kernel     <<<64, 256, 0, stream>>>(Wkv, bkv, query, wqT, cvec);
    score_kernel       <<<256, 256, 0, stream>>>(x, wqT, cvec, sbuf);
    softmax_kernel     <<<BATCH * NHEAD, 256, 0, stream>>>(sbuf, mask, abuf);
    weighted_sum_kernel<<<512, 256, 0, stream>>>(x, abuf, ybuf);
    sampled_kernel     <<<BATCH * NHEAD, 256, 0, stream>>>(ybuf, Wkv, bkv, samp);
    mlp1_kernel        <<<BATCH * 16, 256, 0, stream>>>(samp, W1, b1, hbuf);
    mlp2_kernel        <<<BATCH * 4, 256, 0, stream>>>(hbuf, W2, b2, samp, out);

    (void)in_sizes; (void)n_in; (void)out_size; (void)ws_size;
}
